// Attention_Rel_Scl_5643587027257
// MI455X (gfx1250) — compile-verified, matrix operand order checked by reading, not run on hardware
//
#include <hip/hip_runtime.h>
#include <hip/hip_bf16.h>
#include <math.h>

#define B_  2
#define S_  2048
#define D_  1024
#define H_  16
#define HD_ 64

typedef __attribute__((ext_vector_type(16))) __bf16 v16bf;
typedef __attribute__((ext_vector_type(4)))  __bf16 v4bf;
typedef __attribute__((ext_vector_type(8)))  float  v8f;
typedef __attribute__((ext_vector_type(4)))  unsigned int v4u_t;
typedef __attribute__((ext_vector_type(8)))  int v8i_t;
typedef __attribute__((ext_vector_type(4)))  int v4i_t;

#if defined(__AMDGCN__) && __has_builtin(__builtin_amdgcn_tensor_load_to_lds)
#define HAVE_TDM 1
#else
#define HAVE_TDM 0
#endif

#define APAD 40  // 32 + 8 pad: 80B row stride, 16B aligned, bank-conflict-free
#define KPAD 72  // 64 + 8 pad: 144B stride, 16B aligned, conflict-free

// ---- Tensor Data Mover helpers (CDNA5 ISA) ----
#if HAVE_TDM
// D# group1 for a 2-D bf16 tensor tile with LDS row padding.
// padint: DWORDs-before-pad code (3 => 16 dw, 4 => 32 dw); padamt: 3 => 4 dw pad.
static __device__ __forceinline__ v8i_t tdm_desc2d(uint32_t dim0, uint32_t dim1,
                                                   uint32_t tile0, uint32_t tile1,
                                                   uint32_t stride0,
                                                   uint32_t padint, uint32_t padamt) {
  v8i_t g;
  g[0] = (int)((1u << 16) | (1u << 20) | (padint << 22) | (padamt << 25)); // data_size=2B, pad_enable
  g[1] = (int)((dim0 & 0xFFFFu) << 16);                                   // tensor_dim0[15:0] @ bits 63:48
  g[2] = (int)(((dim0 >> 16) & 0xFFFFu) | ((dim1 & 0xFFFFu) << 16));
  g[3] = (int)(((dim1 >> 16) & 0xFFFFu) | ((tile0 & 0xFFFFu) << 16));     // tile_dim0 @ 127:112
  g[4] = (int)(tile1 & 0xFFFFu);                                          // tile_dim1; tile_dim2=0
  g[5] = (int)stride0;                                                    // tensor_dim0_stride[31:0]
  g[6] = 0;
  g[7] = 0;
  return g;
}

static __device__ __forceinline__ void tdm_issue(uint32_t lds_addr, const void* gaddr, v8i_t g1) {
  uint64_t ga = (uint64_t)(uintptr_t)gaddr;
  v4u_t g0;
  g0[0] = 1u;                                               // count=1, user descriptor
  g0[1] = lds_addr;                                         // LDS byte address
  g0[2] = (uint32_t)ga;                                     // global_addr[31:0] @ bits 95:64
  g0[3] = (uint32_t)((ga >> 32) & 0x1FFFFFFu) | (2u << 30); // global_addr[56:32] | type=2
  v4i_t z4 = {0, 0, 0, 0};
#if __clang_major__ >= 23
  v8i_t z8 = {0, 0, 0, 0, 0, 0, 0, 0};
  __builtin_amdgcn_tensor_load_to_lds(g0, g1, z4, z4, z8, 0);
#else
  __builtin_amdgcn_tensor_load_to_lds(g0, g1, z4, z4, 0);
#endif
}
#endif  // HAVE_TDM

// K index of the low element of VGPR-dword v (0..7) for 16-bit A/B fragments
// (CDNA5 ISA 7.12.2: halves of the wave hold K offsets +8 / +24)
static __device__ __forceinline__ int kpair(int v, int half) {
  return (v < 4) ? (half * 8 + v * 2) : (16 + half * 8 + (v - 4) * 2);
}

// 16x32 bf16 A/B fragment from a row-major LDS row (two ds_load_b128)
static __device__ __forceinline__ v16bf frag_rowk(const __bf16* rowp, int half) {
  v16bf f;
#pragma unroll
  for (int v = 0; v < 8; ++v) {
    int k = kpair(v, half);
    f[2 * v]     = rowp[k];
    f[2 * v + 1] = rowp[k + 1];
  }
  return f;
}

static __device__ __forceinline__ float rmax16(float v) {
#pragma unroll
  for (int m = 8; m >= 1; m >>= 1) v = fmaxf(v, __shfl_xor(v, m, 16));
  return v;
}
static __device__ __forceinline__ float rsum16(float v) {
#pragma unroll
  for (int m = 8; m >= 1; m >>= 1) v += __shfl_xor(v, m, 16);
  return v;
}

// ---------------- f32 -> bf16 convert (vectorized) ----------------
__global__ void f2bf_kernel(const float4* __restrict__ in, v4bf* __restrict__ out, int n4) {
  int i = blockIdx.x * blockDim.x + threadIdx.x;
  int stride = gridDim.x * blockDim.x;
  for (; i < n4; i += stride) {
    float4 v = in[i];
    v4bf o;
    o[0] = (__bf16)v.x; o[1] = (__bf16)v.y; o[2] = (__bf16)v.z; o[3] = (__bf16)v.w;
    out[i] = o;
  }
}

// 8 WMMAs of one k-step: wave computes 32 rows x 64 cols
static __device__ __forceinline__ void gemm_step(const __bf16 (*Asb)[APAD],
                                                 const __bf16 (*Bsb)[APAD],
                                                 int wave, int lr, int half,
                                                 v8f* acc0, v8f* acc1) {
  v16bf a0 = frag_rowk(&Asb[wave * 32 + lr][0], half);
  v16bf a1 = frag_rowk(&Asb[wave * 32 + 16 + lr][0], half);
#pragma unroll
  for (int n = 0; n < 4; ++n) {
    v16bf b = frag_rowk(&Bsb[n * 16 + lr][0], half);
    acc0[n] = __builtin_amdgcn_wmma_f32_16x16x32_bf16(false, a0, false, b,
                                                      (short)0, acc0[n], false, false);
    acc1[n] = __builtin_amdgcn_wmma_f32_16x16x32_bf16(false, a1, false, b,
                                                      (short)0, acc1[n], false, false);
  }
}

// ---------------- QKV projection GEMM: OUT[b,h,s,hd] = sum_d X[b,s,d] * W[e,d] ----------------
// 128 threads (4 waves); 128x64 output tile; TDM double-buffered tile staging.
__global__ __launch_bounds__(128) void gemm_bf16_kernel(const __bf16* __restrict__ X,
                                                        const __bf16* __restrict__ W,
                                                        __bf16* __restrict__ OUT) {
#if HAVE_TDM
  __shared__ __align__(16) __bf16 As[2][128][APAD];
  __shared__ __align__(16) __bf16 Bs[2][64][APAD];
#else
  __shared__ __align__(16) __bf16 As[1][128][APAD];
  __shared__ __align__(16) __bf16 Bs[1][64][APAD];
#endif
  const int bid  = blockIdx.x;
  const int tn   = bid & 15;      // D_/64 = 16 column tiles (== head index)
  const int tm   = bid >> 4;      // (B_*S_)/128 = 32 row tiles
  const int row0 = tm * 128;
  const int col0 = tn * 64;
  const int tid  = threadIdx.x;
  const int lane = tid & 31;
  const int wave = tid >> 5;
  const int half = lane >> 4;
  const int lr   = lane & 15;

  v8f acc0[4] = {};
  v8f acc1[4] = {};

#if HAVE_TDM
  // 2-D tile descriptors: rows of 32 bf16 (16 dw) + 4 dw LDS pad -> APAD rows
  const v8i_t g1A = tdm_desc2d(D_, B_ * S_, 32, 128, D_, /*padint=*/3, /*padamt=*/3);
  const v8i_t g1B = tdm_desc2d(D_, D_,      32, 64,  D_, /*padint=*/3, /*padamt=*/3);
  if (wave == 0) {
    tdm_issue((uint32_t)(uintptr_t)&As[0][0][0], X + (size_t)row0 * D_, g1A);
    tdm_issue((uint32_t)(uintptr_t)&Bs[0][0][0], W + (size_t)col0 * D_, g1B);
  }
  for (int it = 0; it < D_ / 32; ++it) {
    const int k0  = it * 32;
    const int buf = it & 1;
    if (wave == 0) {
      if (k0 + 32 < D_) {
        tdm_issue((uint32_t)(uintptr_t)&As[buf ^ 1][0][0], X + (size_t)row0 * D_ + k0 + 32, g1A);
        tdm_issue((uint32_t)(uintptr_t)&Bs[buf ^ 1][0][0], W + (size_t)col0 * D_ + k0 + 32, g1B);
        __builtin_amdgcn_s_wait_tensorcnt(2);   // current pair done; next pair in flight
      } else {
        __builtin_amdgcn_s_wait_tensorcnt(0);
      }
    }
    __syncthreads();
    gemm_step(As[buf], Bs[buf], wave, lr, half, acc0, acc1);
    __syncthreads();   // all reads of buf done before it is overwritten next-next step
  }
#else
  for (int k0 = 0; k0 < D_; k0 += 32) {
    __syncthreads();
    for (int i = tid; i < 512; i += 128) {
      int r = i >> 2, c = i & 3;
      *(uint4*)&As[0][r][c * 8] =
          *(const uint4*)(X + (size_t)(row0 + r) * D_ + k0 + c * 8);
    }
    for (int i = tid; i < 256; i += 128) {
      int r = i >> 2, c = i & 3;
      *(uint4*)&Bs[0][r][c * 8] =
          *(const uint4*)(W + (size_t)(col0 + r) * D_ + k0 + c * 8);
    }
    __syncthreads();
    gemm_step(As[0], Bs[0], wave, lr, half, acc0, acc1);
  }
#endif

  const int h = tn;
#pragma unroll
  for (int strip = 0; strip < 2; ++strip) {
#pragma unroll
    for (int n = 0; n < 4; ++n) {
#pragma unroll
      for (int r = 0; r < 8; ++r) {
        float val = strip ? acc1[n][r] : acc0[n][r];
        int m = wave * 32 + strip * 16 + r + 8 * half;
        size_t grow = (size_t)row0 + m;
        int b = (int)(grow / S_);
        int s = (int)(grow % S_);
        int hd = n * 16 + lr;
        OUT[(((size_t)b * H_ + h) * S_ + s) * HD_ + hd] = (__bf16)val;
      }
    }
  }
}

// ---------------- fused attention + post-softmax relative bias ----------------
// 4 waves share (b,h); each wave owns a 16-row query strip; online softmax.
__global__ __launch_bounds__(128) void attn_kernel(const __bf16* __restrict__ Q,
                                                   const __bf16* __restrict__ K,
                                                   const __bf16* __restrict__ V,
                                                   const float* __restrict__ bias_table,
                                                   float* __restrict__ AO) {
  __shared__ __align__(16) __bf16 Ks[32][KPAD];     // [key][hd]
  __shared__ __align__(16) __bf16 Vt[64][APAD];     // [hd][key]  (transposed at load)
  __shared__ __align__(16) __bf16 Ps[4][16][APAD];  // per-wave C->A transpose scratch
  __shared__ __bf16 bias_sh[96];                    // staged bias rows per j-step

  const int bid  = blockIdx.x;
  const int qb   = bid & 31;             // S_/64 = 32
  const int h    = (bid >> 5) & 15;
  const int b    = bid >> 9;
  const int tid  = threadIdx.x;
  const int lane = tid & 31;
  const int wave = tid >> 5;
  const int half = lane >> 4;
  const int lr   = lane & 15;

  const size_t headoff = (((size_t)b * H_ + h) * S_) * HD_;
  const __bf16* Qh = Q + headoff;
  const __bf16* Kh = K + headoff;
  const __bf16* Vh = V + headoff;

  const int q0_blk = qb * 64;
  const int q0     = q0_blk + wave * 16;
  const int qrow   = q0 + lr;

  v16bf qa0 = frag_rowk(Qh + (size_t)qrow * HD_, half);       // hd 0..31
  v16bf qa1 = frag_rowk(Qh + (size_t)qrow * HD_ + 32, half);  // hd 32..63

  float mrow[8], lrow[8], alpha[8];
  v8f accS[4] = {};
  v8f accB[4] = {};
#pragma unroll
  for (int r = 0; r < 8; ++r) { mrow[r] = -1e30f; lrow[r] = 0.f; }

  const float scale = 0.03125f;  // D_^-0.5

#if HAVE_TDM
  // rows of 64 bf16 (32 dw) + 4 dw LDS pad -> KPAD rows
  const v8i_t g1K = tdm_desc2d(HD_, S_, 64, 32, HD_, /*padint=*/4, /*padamt=*/3);
#endif

  for (int j0 = 0; j0 < S_; j0 += 32) {
    __syncthreads();
#if HAVE_TDM
    if (wave == 0)
      tdm_issue((uint32_t)(uintptr_t)&Ks[0][0], Kh + (size_t)j0 * HD_, g1K);
#else
    for (int i = tid; i < 256; i += 128) {
      int r = i >> 3, c = i & 7;
      *(uint4*)&Ks[r][c * 8] =
          *(const uint4*)(Kh + (size_t)(j0 + r) * HD_ + c * 8);
    }
#endif
    // V tile: load 8-bf16 chunks along hd, scatter transposed into Vt[hd][key]
    for (int i = tid; i < 256; i += 128) {
      int r = i >> 3, c = i & 7;                 // key r, hd base c*8
      v4bf lo = *(const v4bf*)(Vh + (size_t)(j0 + r) * HD_ + c * 8);
      v4bf hi = *(const v4bf*)(Vh + (size_t)(j0 + r) * HD_ + c * 8 + 4);
#pragma unroll
      for (int e = 0; e < 4; ++e) {
        Vt[c * 8 + e][r]     = lo[e];
        Vt[c * 8 + 4 + e][r] = hi[e];
      }
    }
    // bias slice: 95 consecutive table rows cover every (q,k) pair of this step
    const int base = q0_blk - j0 - 31 + (S_ - 1);   // always in [0, 2S-2-94]
    if (tid < 96) bias_sh[tid] = (__bf16)bias_table[(size_t)(base + tid) * H_ + h];
#if HAVE_TDM
    if (wave == 0) __builtin_amdgcn_s_wait_tensorcnt(0);
#endif
    __syncthreads();

    // prefetch next K/V tiles into cache while we compute (global_prefetch_b8)
    if (j0 + 32 < S_) {
      __builtin_prefetch(Kh + (size_t)(j0 + 32) * HD_ + tid * 16, 0, 0);
      __builtin_prefetch(Vh + (size_t)(j0 + 32) * HD_ + tid * 16, 0, 0);
    }

    // scores: two 16x16 tiles over the 32-key step (K-dim = HD = 2x32)
    v8f sc[2];
#pragma unroll
    for (int t = 0; t < 2; ++t) {
      v8f s = {};
      v16bf kb0 = frag_rowk(&Ks[t * 16 + lr][0], half);
      s = __builtin_amdgcn_wmma_f32_16x16x32_bf16(false, qa0, false, kb0, (short)0, s, false, false);
      v16bf kb1 = frag_rowk(&Ks[t * 16 + lr][32], half);
      s = __builtin_amdgcn_wmma_f32_16x16x32_bf16(false, qa1, false, kb1, (short)0, s, false, false);
      sc[t] = s;
    }

    // online softmax (row = r + 8*half; its 16 cols live in this 16-lane half)
#pragma unroll
    for (int r = 0; r < 8; ++r) {
      float v0 = sc[0][r] * scale;
      float v1 = sc[1][r] * scale;
      float tmax = rmax16(fmaxf(v0, v1));
      float mnew = fmaxf(mrow[r], tmax);
      alpha[r] = __expf(mrow[r] - mnew);
      float p0 = __expf(v0 - mnew);
      float p1 = __expf(v1 - mnew);
      lrow[r] = lrow[r] * alpha[r] + rsum16(p0 + p1);
      mrow[r] = mnew;
      int m = r + 8 * half;
      Ps[wave][m][lr]      = (__bf16)p0;        // C-layout -> A-layout via LDS
      Ps[wave][m][16 + lr] = (__bf16)p1;
    }
#pragma unroll
    for (int n = 0; n < 4; ++n)
#pragma unroll
      for (int r = 0; r < 8; ++r) accS[n][r] *= alpha[r];

    v16bf pf = frag_rowk(&Ps[wave][lr][0], half);

    // relative-bias A-fragment from staged LDS slice:
    //   bias[q, j0+k] = bias_sh[wave*16 + lr + 31 - k]
    v16bf bfF;
#pragma unroll
    for (int v = 0; v < 8; ++v) {
      int k   = kpair(v, half);
      int off = wave * 16 + lr + 31 - k;
      bfF[2 * v]     = bias_sh[off];
      bfF[2 * v + 1] = bias_sh[off - 1];
    }

#pragma unroll
    for (int n = 0; n < 4; ++n) {
      // B-fragment of V from transposed tile: row = hd, contiguous K = key
      v16bf vf = frag_rowk(&Vt[n * 16 + lr][0], half);
      accS[n] = __builtin_amdgcn_wmma_f32_16x16x32_bf16(false, pf,  false, vf, (short)0, accS[n], false, false);
      accB[n] = __builtin_amdgcn_wmma_f32_16x16x32_bf16(false, bfF, false, vf, (short)0, accB[n], false, false);
    }
  }

  // out = softmax(S)V / l + (Bias)V, back to [B,S,D] (heads recombined)
#pragma unroll
  for (int n = 0; n < 4; ++n) {
#pragma unroll
    for (int r = 0; r < 8; ++r) {
      int m  = r + 8 * half;
      int qg = q0 + m;
      int d  = h * HD_ + n * 16 + lr;
      float val = accS[n][r] / lrow[r] + accB[n][r];
      AO[((size_t)b * S_ + qg) * D_ + d] = val;
    }
  }
}

// ---------------- LayerNorm over D ----------------
__global__ __launch_bounds__(256) void ln_kernel(const float* __restrict__ X,
                                                 const float* __restrict__ g,
                                                 const float* __restrict__ be,
                                                 float* __restrict__ out) {
  __shared__ float shs[8], shq[8], stats[2];
  const int row = blockIdx.x;
  const float* p = X + (size_t)row * D_;
  float s = 0.f, q = 0.f;
  for (int i = threadIdx.x; i < D_; i += 256) {
    float v = p[i];
    s += v; q += v * v;
  }
#pragma unroll
  for (int m = 16; m >= 1; m >>= 1) {
    s += __shfl_xor(s, m, 32);
    q += __shfl_xor(q, m, 32);
  }
  int wave = threadIdx.x >> 5;
  if ((threadIdx.x & 31) == 0) { shs[wave] = s; shq[wave] = q; }
  __syncthreads();
  if (threadIdx.x == 0) {
    float S = 0.f, Q = 0.f;
    for (int w = 0; w < 8; ++w) { S += shs[w]; Q += shq[w]; }
    float mu  = S / (float)D_;
    float var = Q / (float)D_ - mu * mu;
    stats[0] = mu;
    stats[1] = rsqrtf(var + 1e-5f);
  }
  __syncthreads();
  float mu = stats[0], rs = stats[1];
  for (int i = threadIdx.x; i < D_; i += 256) {
    out[(size_t)row * D_ + i] = (p[i] - mu) * rs * g[i] + be[i];
  }
}

extern "C" void kernel_launch(void* const* d_in, const int* in_sizes, int n_in,
                              void* d_out, int out_size, void* d_ws, size_t ws_size,
                              hipStream_t stream) {
  (void)in_sizes; (void)n_in; (void)out_size; (void)ws_size;
  const float* x  = (const float*)d_in[0];
  const float* Wq = (const float*)d_in[1];
  const float* Wk = (const float*)d_in[2];
  const float* Wv = (const float*)d_in[3];
  const float* bt = (const float*)d_in[4];
  const float* g  = (const float*)d_in[5];
  const float* be = (const float*)d_in[6];

  char* ws = (char*)d_ws;
  const size_t MB = (size_t)1 << 20;
  __bf16* xb  = (__bf16*)(ws);              //  8 MB  x in bf16
  __bf16* wqb = (__bf16*)(ws +  8 * MB);    //  2 MB
  __bf16* wkb = (__bf16*)(ws + 10 * MB);    //  2 MB
  __bf16* wvb = (__bf16*)(ws + 12 * MB);    //  2 MB
  __bf16* Qh  = (__bf16*)(ws + 14 * MB);    //  8 MB  [B,H,S,HD]
  __bf16* Kh  = (__bf16*)(ws + 22 * MB);    //  8 MB
  __bf16* Vh  = (__bf16*)(ws + 30 * MB);    //  8 MB
  float*  AO  = (float*) (ws + 38 * MB);    // 16 MB  pre-LN output [B,S,D]

  const int NX = B_ * S_ * D_;   // 4194304
  const int NW = D_ * D_;        // 1048576
  f2bf_kernel<<<1024, 256, 0, stream>>>((const float4*)x,  (v4bf*)xb,  NX / 4);
  f2bf_kernel<<<512,  256, 0, stream>>>((const float4*)Wq, (v4bf*)wqb, NW / 4);
  f2bf_kernel<<<512,  256, 0, stream>>>((const float4*)Wk, (v4bf*)wkb, NW / 4);
  f2bf_kernel<<<512,  256, 0, stream>>>((const float4*)Wv, (v4bf*)wvb, NW / 4);

  gemm_bf16_kernel<<<512, 128, 0, stream>>>(xb, wqb, Qh);
  gemm_bf16_kernel<<<512, 128, 0, stream>>>(xb, wkb, Kh);
  gemm_bf16_kernel<<<512, 128, 0, stream>>>(xb, wvb, Vh);

  attn_kernel<<<B_ * H_ * (S_ / 64), 128, 0, stream>>>(Qh, Kh, Vh, bt, AO);

  ln_kernel<<<B_ * S_, 256, 0, stream>>>(AO, g, be, (float*)d_out);
}
